// SwitchHeadAttention_29240137351327
// MI455X (gfx1250) — matrix operand order repaired and checked by reading, not compiled, not run on hardware
//
#include <hip/hip_runtime.h>
#include <hip/hip_bf16.h>

typedef __attribute__((ext_vector_type(16))) _Float16 v16h;
typedef __attribute__((ext_vector_type(8)))  _Float16 v8h;
typedef __attribute__((ext_vector_type(8)))  float    v8f;

namespace sh {
constexpr int B  = 4;
constexpr int T  = 2048;
constexpr int D  = 1024;
constexpr int H  = 8;
constexpr int DH = 64;
constexpr int E  = 5;
constexpr int M  = B * T;    // 8192 tokens
constexpr int NQ = H * DH;   // 512
constexpr int NE = E * DH;   // 320
constexpr int G  = H * E;    // 40 gate columns
}
using namespace sh;

// ---------------- helpers ----------------
__device__ __forceinline__ v16h cat16(v8h a, v8h b) {
  return __builtin_shufflevector(a, b, 0,1,2,3,4,5,6,7,8,9,10,11,12,13,14,15);
}
__device__ __forceinline__ v8f wmma32(v16h a, v16h b, v8f c) {
  // D = A(16x32 f16) * B(32x16 f16) + C(16x16 f32)
  return __builtin_amdgcn_wmma_f32_16x16x32_f16(false, a, false, b, (short)0, c, false, false);
}
// row-wise reductions over the 16-lane halves of a wave32 (rows live across lanes in C layout)
__device__ __forceinline__ float redmax16(float v) {
  v = fmaxf(v, __shfl_xor(v, 1, 32));
  v = fmaxf(v, __shfl_xor(v, 2, 32));
  v = fmaxf(v, __shfl_xor(v, 4, 32));
  v = fmaxf(v, __shfl_xor(v, 8, 32));
  return v;
}
__device__ __forceinline__ float redsum16(float v) {
  v += __shfl_xor(v, 1, 32);
  v += __shfl_xor(v, 2, 32);
  v += __shfl_xor(v, 4, 32);
  v += __shfl_xor(v, 8, 32);
  return v;
}

// ---------------- prep / repack ----------------
__global__ void k_cvt_f16(const float* __restrict__ in, _Float16* __restrict__ out, int n) {
  for (int i = blockIdx.x * blockDim.x + threadIdx.x; i < n; i += gridDim.x * blockDim.x)
    out[i] = (_Float16)in[i];
}

// W (K x N) row-major f32 -> Wt (N x K) row-major f16
__global__ void k_transpose_f16(const float* __restrict__ W, _Float16* __restrict__ Wt,
                                int K, int N) {
  int n = K * N;
  for (int i = blockIdx.x * blockDim.x + threadIdx.x; i < n; i += gridDim.x * blockDim.x) {
    int r = i / K, c = i % K;                 // out row = orig col, out col = orig row
    Wt[i] = (_Float16)W[(size_t)c * N + r];
  }
}

// Wv (E, D, DH) -> Wv_t (E*DH, D) f16   (row r = e*DH+kk holds Wv[e, :, kk])
__global__ void k_repack_wv(const float* __restrict__ Wv, _Float16* __restrict__ Wt) {
  int n = NE * D;
  for (int i = blockIdx.x * blockDim.x + threadIdx.x; i < n; i += gridDim.x * blockDim.x) {
    int r = i / D, d = i % D;
    int e = r >> 6, kk = r & 63;
    Wt[i] = (_Float16)Wv[((size_t)e * D + d) * DH + kk];
  }
}

// Wo (E*DH, D) -> Wo_t (D, E*DH) f16
__global__ void k_repack_wo(const float* __restrict__ Wo, _Float16* __restrict__ Wt) {
  int n = D * NE;
  for (int i = blockIdx.x * blockDim.x + threadIdx.x; i < n; i += gridDim.x * blockDim.x) {
    int d = i / NE, r = i % NE;
    Wt[i] = (_Float16)Wo[(size_t)r * D + d];
  }
}

// ---------------- gates (N=40 each, scalar path; x row staged in LDS) ----------------
__global__ __launch_bounds__(128) void k_gates(const float* __restrict__ x,
                                               const float* __restrict__ Ws,
                                               const float* __restrict__ Wd,
                                               float* __restrict__ glv,
                                               float* __restrict__ glo) {
  __shared__ float xs[D];
  int m = blockIdx.x, tid = threadIdx.x;
  for (int i = tid; i < D; i += 128) xs[i] = x[(size_t)m * D + i];
  __syncthreads();
  if (tid < 2 * G) {
    const float* W = (tid < G) ? Ws : Wd;
    int nn = (tid < G) ? tid : tid - G;
    float a = 0.f;
    for (int kk = 0; kk < D; ++kk) a = fmaf(xs[kk], W[(size_t)kk * G + nn], a);
    ((tid < G) ? glv : glo)[(size_t)m * G + nn] = a;
  }
}

// per-(token,head): top-2 of 5 -> wv = mask*sigmoid(gl_v), mo = mask
__global__ __launch_bounds__(32) void k_gate_weights(const float* __restrict__ glv,
                                                     const float* __restrict__ glo,
                                                     float* __restrict__ wvw,
                                                     float* __restrict__ mow) {
  int m = blockIdx.x, h = threadIdx.x;
  if (h >= H) return;
  const float* gv = glv + (size_t)m * G + h * E;
  const float* go = glo + (size_t)m * G + h * E;
  float* ov = wvw + (size_t)m * G + h * E;
  float* oo = mow + (size_t)m * G + h * E;
  int i1 = 0;
  for (int e = 1; e < E; ++e) if (gv[e] > gv[i1]) i1 = e;
  int i2 = (i1 == 0) ? 1 : 0;
  for (int e = 0; e < E; ++e) if (e != i1 && gv[e] > gv[i2]) i2 = e;
  for (int e = 0; e < E; ++e) {
    float s = 1.f / (1.f + expf(-gv[e]));
    ov[e] = (e == i1 || e == i2) ? s : 0.f;
  }
  int j1 = 0;
  for (int e = 1; e < E; ++e) if (go[e] > go[j1]) j1 = e;
  int j2 = (j1 == 0) ? 1 : 0;
  for (int e = 0; e < E; ++e) if (e != j1 && go[e] > go[j2]) j2 = e;
  for (int e = 0; e < E; ++e) oo[e] = (e == j1 || e == j2) ? 1.f : 0.f;
}

// ---------------- WMMA GEMM: C(MxN) = A(MxK f16) * Bt(NxK f16)^T ----------------
// one wave per 16x64 output tile (4 accumulators, A fragment reused 4x)
// mode 0: f32 linear out[m*N + n] * scale
// mode 1: f16 head layout out[((b*H+h)*T+t)*DH+dh] * scale   (N must be H*DH)
__global__ __launch_bounds__(32) void k_gemm(const _Float16* __restrict__ A,
                                             const _Float16* __restrict__ Bt,
                                             float* __restrict__ outF,
                                             _Float16* __restrict__ outH,
                                             int Kdim, int Ndim, float scale, int mode) {
  const int lane = threadIdx.x & 31;
  const int lx = lane & 15, hg = lane >> 4;
  const int m0 = blockIdx.y << 4;
  const int n0 = blockIdx.x << 6;
  v8f acc[4] = {};
  const _Float16* arow = A + (size_t)(m0 + lx) * Kdim;
  const _Float16* brow = Bt + (size_t)(n0 + lx) * Kdim;
  for (int k = 0; k < Kdim; k += 32) {
    const _Float16* ap = arow + k + hg * 16;
    v16h af = cat16(*(const v8h*)ap, *(const v8h*)(ap + 8));
    __builtin_prefetch(ap + 32, 0, 3);           // global_prefetch_b8 on next A chunk
    const _Float16* bp = brow + k + hg * 16;
    acc[0] = wmma32(af, *(const v16h*)(bp),                       acc[0]);
    acc[1] = wmma32(af, *(const v16h*)(bp + (size_t)16 * Kdim),   acc[1]);
    acc[2] = wmma32(af, *(const v16h*)(bp + (size_t)32 * Kdim),   acc[2]);
    acc[3] = wmma32(af, *(const v16h*)(bp + (size_t)48 * Kdim),   acc[3]);
  }
#pragma unroll
  for (int s = 0; s < 4; ++s) {
#pragma unroll
    for (int i = 0; i < 8; ++i) {
      int mrow = m0 + (hg ? i + 8 : i);
      int n = n0 + s * 16 + lx;
      float v = acc[s][i] * scale;
      if (mode == 0) {
        outF[(size_t)mrow * Ndim + n] = v;
      } else {
        int bb = mrow / T, tt = mrow % T;
        int hh = n >> 6, dd = n & 63;
        outH[(((size_t)bb * H + hh) * T + tt) * DH + dd] = (_Float16)v;
      }
    }
  }
}

// ---------------- V combine: Vt[b,h,dh,t] = sum_e wv * xv  (transposed for PV B-frags) ----
__global__ __launch_bounds__(512) void k_combine_v(const float* __restrict__ xv,
                                                   const float* __restrict__ wvw,
                                                   _Float16* __restrict__ Vt) {
  __shared__ float xs[NE];
  __shared__ float ws2[G];
  int m = blockIdx.x, tid = threadIdx.x;
  if (tid < NE) xs[tid] = xv[(size_t)m * NE + tid];
  else if (tid < NE + G) ws2[tid - NE] = wvw[(size_t)m * G + (tid - NE)];
  __syncthreads();
  int h = tid >> 6, kk = tid & 63;
  float s = 0.f;
#pragma unroll
  for (int e = 0; e < E; ++e) s = fmaf(ws2[h * E + e], xs[e * DH + kk], s);
  int bb = m / T, tt = m % T;
  Vt[(((size_t)bb * H + h) * DH + kk) * T + tt] = (_Float16)s;
}

// ---------------- flash attention: 1 wave = 16 queries of one (b,h); 32-key blocks --------
// Q is pre-scaled by dh^-0.5 * log2(e), so softmax runs in exp2 domain.
__global__ __launch_bounds__(32) void k_flash(const _Float16* __restrict__ Q,
                                              const _Float16* __restrict__ Kh,
                                              const _Float16* __restrict__ Vt,
                                              float* __restrict__ attn) {
  const int lane = threadIdx.x & 31;
  const int lx = lane & 15, hg = lane >> 4;
  const int q0 = blockIdx.x << 4;
  const int bh = blockIdx.y;
  const int bb = bh / H, hh = bh % H;
  const _Float16* Qb = Q  + (size_t)bh * T * DH;
  const _Float16* Kb = Kh + (size_t)bh * T * DH;
  const _Float16* Vb = Vt + (size_t)bh * DH * T;
  __shared__ __align__(32) _Float16 plds[16 * 32];   // P tile bounce: C-layout -> A-layout

  v16h qf0, qf1;
  {
    const _Float16* qp = Qb + (size_t)(q0 + lx) * DH + hg * 16;
    qf0 = cat16(*(const v8h*)qp,        *(const v8h*)(qp + 8));
    qf1 = cat16(*(const v8h*)(qp + 32), *(const v8h*)(qp + 40));
  }
  v8f o0 = {}, o1 = {}, o2 = {}, o3 = {};
  float mi[8], li[8];
#pragma unroll
  for (int i = 0; i < 8; ++i) { mi[i] = -1e30f; li[i] = 0.f; }

  for (int j = 0; j < T; j += 32) {
    // S = Q * K^T for 32 keys (two 16x16 N-tiles, K-dim = DH = 2x32)
    v8f c0 = {}, c1 = {};
    {
      const int kd = hg * 16;
      const _Float16* kp0 = Kb + (size_t)(j + lx) * DH + kd;
      const _Float16* kp1 = Kb + (size_t)(j + 16 + lx) * DH + kd;
      c0 = wmma32(qf0, *(const v16h*)(kp0),      c0);
      c0 = wmma32(qf1, *(const v16h*)(kp0 + 32), c0);
      c1 = wmma32(qf0, *(const v16h*)(kp1),      c1);
      c1 = wmma32(qf1, *(const v16h*)(kp1 + 32), c1);
    }
    // online softmax (per C-layout element = per query row)
    float alpha[8];
#pragma unroll
    for (int i = 0; i < 8; ++i) {
      float mx = redmax16(fmaxf(c0[i], c1[i]));
      float mn = fmaxf(mi[i], mx);
      float p0 = exp2f(c0[i] - mn);
      float p1 = exp2f(c1[i] - mn);
      float rs = redsum16(p0 + p1);
      float al = exp2f(mi[i] - mn);
      li[i] = li[i] * al + rs;
      mi[i] = mn;
      alpha[i] = al;
      int r = hg ? i + 8 : i;
      plds[r * 32 + lx]      = (_Float16)p0;   // P column = key offset 0..15
      plds[r * 32 + 16 + lx] = (_Float16)p1;   // P column = key offset 16..31
    }
#pragma unroll
    for (int i = 0; i < 8; ++i) {
      o0[i] *= alpha[i]; o1[i] *= alpha[i]; o2[i] *= alpha[i]; o3[i] *= alpha[i];
    }
    // reload P as an A fragment (row per lane, two 8-half chunks); same-wave DS ops are in-order
    v16h pf;
    {
      const _Float16* pp = plds + lx * 32 + hg * 16;
      pf = cat16(*(const v8h*)pp, *(const v8h*)(pp + 8));
    }
    // O += P(16x32) * V(32keys x 64dh): V stored (dh, t) so B-frags are contiguous
    const int kb = j + hg * 16;
    o0 = wmma32(pf, *(const v16h*)(Vb + (size_t)(lx)      * T + kb), o0);
    o1 = wmma32(pf, *(const v16h*)(Vb + (size_t)(16 + lx) * T + kb), o1);
    o2 = wmma32(pf, *(const v16h*)(Vb + (size_t)(32 + lx) * T + kb), o2);
    o3 = wmma32(pf, *(const v16h*)(Vb + (size_t)(48 + lx) * T + kb), o3);
  }
#pragma unroll
  for (int i = 0; i < 8; ++i) {
    int r = hg ? i + 8 : i;
    int tt = q0 + r;
    float inv = 1.0f / li[i];
    float* dst = attn + (((size_t)bb * T + tt) * H + hh) * DH + lx;
    dst[0]  = o0[i] * inv;
    dst[16] = o1[i] * inv;
    dst[32] = o2[i] * inv;
    dst[48] = o3[i] * inv;
  }
}

// ---------------- C combine: Ct[m, e*DH+k] = sum_h mask_o * attn  (f16 for final GEMM) ----
__global__ __launch_bounds__(512) void k_combine_c(const float* __restrict__ attn,
                                                   const float* __restrict__ mow,
                                                   _Float16* __restrict__ Ct) {
  __shared__ float as[NQ];
  __shared__ float ms[G];
  int m = blockIdx.x, tid = threadIdx.x;
  as[tid] = attn[(size_t)m * NQ + tid];
  if (tid < G) ms[tid] = mow[(size_t)m * G + tid];
  __syncthreads();
  if (tid < NE) {
    int e = tid >> 6, kk = tid & 63;
    float s = 0.f;
#pragma unroll
    for (int h2 = 0; h2 < H; ++h2) s = fmaf(ms[h2 * E + e], as[h2 * DH + kk], s);
    Ct[(size_t)m * NE + tid] = (_Float16)s;
  }
}

// ---------------- launcher ----------------
extern "C" void kernel_launch(void* const* d_in, const int* in_sizes, int n_in,
                              void* d_out, int out_size, void* d_ws, size_t ws_size,
                              hipStream_t stream) {
  (void)in_sizes; (void)n_in; (void)out_size; (void)ws_size;
  const float* x  = (const float*)d_in[0];
  const float* Wq = (const float*)d_in[1];
  const float* Wk = (const float*)d_in[2];
  const float* Ws = (const float*)d_in[3];
  const float* Wd = (const float*)d_in[4];
  const float* Wv = (const float*)d_in[5];
  const float* Wo = (const float*)d_in[6];
  float* out = (float*)d_out;

  char* base = (char*)d_ws;
  size_t off = 0;
  auto carve = [&](size_t bytes) -> void* {
    void* p = base + off;
    off = (off + bytes + 255) & ~(size_t)255;
    return p;
  };
  _Float16* xf  = (_Float16*)carve((size_t)M * D * 2);    // x in f16
  _Float16* wqT = (_Float16*)carve((size_t)NQ * D * 2);   // Wq^T (N,K) f16
  _Float16* wkT = (_Float16*)carve((size_t)NQ * D * 2);
  _Float16* wvT = (_Float16*)carve((size_t)NE * D * 2);   // Wv repacked (E*DH, D)
  _Float16* woT = (_Float16*)carve((size_t)D * NE * 2);   // Wo^T (D, E*DH)
  _Float16* Qh  = (_Float16*)carve((size_t)M * NQ * 2);   // (b,h,t,dh) pre-scaled
  _Float16* Kh  = (_Float16*)carve((size_t)M * NQ * 2);   // (b,h,t,dh)
  _Float16* Vt  = (_Float16*)carve((size_t)M * NQ * 2);   // (b,h,dh,t)
  float* glv  = (float*)carve((size_t)M * G * 4);
  float* glo  = (float*)carve((size_t)M * G * 4);
  float* wvw  = (float*)carve((size_t)M * G * 4);
  float* mow  = (float*)carve((size_t)M * G * 4);
  float* xv   = (float*)carve((size_t)M * NE * 4);        // all-expert V projections
  float* attn = (float*)carve((size_t)M * NQ * 4);        // (b,t,h,dh)
  _Float16* Ct = (_Float16*)carve((size_t)M * NE * 2);    // masked head-sum, f16

  const float QSCALE = 0.125f * 1.44269504088896340736f;  // dh^-0.5 * log2(e)

  // prep + repack
  k_cvt_f16<<<4096, 256, 0, stream>>>(x, xf, M * D);
  k_transpose_f16<<<1024, 256, 0, stream>>>(Wq, wqT, D, NQ);
  k_transpose_f16<<<1024, 256, 0, stream>>>(Wk, wkT, D, NQ);
  k_repack_wv<<<1024, 256, 0, stream>>>(Wv, wvT);
  k_repack_wo<<<1024, 256, 0, stream>>>(Wo, woT);

  // gating
  k_gates<<<M, 128, 0, stream>>>(x, Ws, Wd, glv, glo);
  k_gate_weights<<<M, 32, 0, stream>>>(glv, glo, wvw, mow);

  // projections (WMMA)
  k_gemm<<<dim3(NQ / 64, M / 16), 32, 0, stream>>>(xf, wqT, nullptr, Qh, D, NQ, QSCALE, 1);
  k_gemm<<<dim3(NQ / 64, M / 16), 32, 0, stream>>>(xf, wkT, nullptr, Kh, D, NQ, 1.0f,  1);
  k_gemm<<<dim3(NE / 64, M / 16), 32, 0, stream>>>(xf, wvT, xv,    nullptr, D, NE, 1.0f, 0);
  k_combine_v<<<M, 512, 0, stream>>>(xv, wvw, Vt);

  // attention (WMMA flash)
  k_flash<<<dim3(T / 16, B * H), 32, 0, stream>>>(Qh, Kh, Vt, attn);

  // MoE output: C = mask-combined heads, then dense GEMM with flattened Wo (WMMA)
  k_combine_c<<<M, 512, 0, stream>>>(attn, mow, Ct);
  k_gemm<<<dim3(D / 64, M / 16), 32, 0, stream>>>(Ct, woT, out, nullptr, NE, D, 1.0f, 0);
}
